// AxialAttention_32152125177891
// MI455X (gfx1250) — matrix operand order repaired and checked by reading, not compiled, not run on hardware
//
#include <hip/hip_runtime.h>
#include <hip/hip_bf16.h>
#include <stdint.h>

#define N_   896
#define C_   1024
#define L_   56
#define H_   8
#define KD_  512
#define VD_  1024
#define DK_  64
#define DV_  128
#define OCH  (2*KD_ + VD_)   // 2048
#define SIMC (3*H_)          // 24
#define SPAN 255
#define EPS_ 1e-3f

typedef __attribute__((ext_vector_type(16))) __bf16 v16bf;
typedef __attribute__((ext_vector_type(8)))  float  v8f;

// ---------------- f32 -> bf16 conversion ----------------
__global__ void cvt_bf16(const float* __restrict__ in, __bf16* __restrict__ out, int n) {
  int i = blockIdx.x * blockDim.x + threadIdx.x;
  if (i < n) out[i] = (__bf16)in[i];
}

__global__ void zero_kernel(float* __restrict__ p, int n) {
  int i = blockIdx.x * blockDim.x + threadIdx.x;
  if (i < n) p[i] = 0.0f;
}

// ---------------- projection GEMM: qkv[b,o,l] = sum_c W[o,c]*x[b,c,l] ----------------
// Block tile: 256(M=o) x 64(N=l, padded from 56) x 32(K=c).
// 8 waves in 4x2 layout; each wave owns 64x32 = 4x2 WMMA tiles (8 WMMA / k-step).
__global__ __launch_bounds__(256) void gemm_qkv(const __bf16* __restrict__ W,
                                                const __bf16* __restrict__ X,
                                                float* __restrict__ QKV) {
  __shared__ __align__(16) __bf16 As[256][32];   // W tile, [m][k]
  __shared__ __align__(16) __bf16 Bs[64][32];    // x tile transposed, [l][k]
  const int b    = blockIdx.y;
  const int m0   = blockIdx.x * 256;
  const int tid  = threadIdx.x;
  const int lane = tid & 31;
  const int wave = tid >> 5;
  const int wm   = (wave >> 1) * 64;   // 0,64,128,192
  const int wn   = (wave & 1) * 32;    // 0,32

  v8f acc[4][2] = {};

  const __bf16* Wp = W + (size_t)m0 * C_;
  const __bf16* Xp = X + (size_t)b * C_ * L_;

  const int bc = tid >> 3;            // B k-local 0..31
  const int bl = (tid & 7) * 8;       // B l base 0..56

  // zero-fill the pad rows (l = 56..63) once; they are never rewritten in-loop
  if (bl >= L_) {
    #pragma unroll
    for (int j = 0; j < 8; ++j) Bs[bl + j][bc] = (__bf16)0.0f;
  }

  for (int k0 = 0; k0 < C_; k0 += 32) {
    // A tile: one full 32-half row per thread
    const __bf16* asrc = Wp + (size_t)tid * C_ + k0;
    *(float4*)&As[tid][0]  = ((const float4*)asrc)[0];
    *(float4*)&As[tid][8]  = ((const float4*)asrc)[1];
    *(float4*)&As[tid][16] = ((const float4*)asrc)[2];
    *(float4*)&As[tid][24] = ((const float4*)asrc)[3];
    // B tile (transposed into Bs[l][k])
    if (bl < L_) {
      union { float4 f; __bf16 h[8]; } u;
      u.f = *(const float4*)(Xp + (size_t)(k0 + bc) * L_ + bl);
      #pragma unroll
      for (int j = 0; j < 8; ++j) Bs[bl + j][bc] = u.h[j];
    }
    __syncthreads();

    union Fr { v16bf v; float4 f[2]; };
    Fr a[4], bf[2];
    const int rsel = lane & 15;
    const int kh = (lane >> 4) * 8;    // A: lanes0-15 -> K0-7/16-23, lanes16-31 -> K8-15/24-31
    const int kb = (lane >> 4) * 8;    // B: lanes0-15 -> K0-7/16-23, lanes16-31 -> K8-15/24-31
    #pragma unroll
    for (int tm = 0; tm < 4; ++tm) {
      int r = wm + tm * 16 + rsel;
      a[tm].f[0] = *(float4*)&As[r][kh];
      a[tm].f[1] = *(float4*)&As[r][16 + kh];
    }
    #pragma unroll
    for (int tn = 0; tn < 2; ++tn) {
      int n = wn + tn * 16 + rsel;
      bf[tn].f[0] = *(float4*)&Bs[n][kb];
      bf[tn].f[1] = *(float4*)&Bs[n][kb + 16];
    }
    #pragma unroll
    for (int tm = 0; tm < 4; ++tm)
      #pragma unroll
      for (int tn = 0; tn < 2; ++tn)
        acc[tm][tn] = __builtin_amdgcn_wmma_f32_16x16x32_bf16(
            false, a[tm].v, false, bf[tn].v, (short)0, acc[tm][tn], false, false);
    __syncthreads();
  }

  const int n  = lane & 15;
  const int mh = (lane >> 4) * 8;
  #pragma unroll
  for (int tm = 0; tm < 4; ++tm) {
    #pragma unroll
    for (int tn = 0; tn < 2; ++tn) {
      int l = wn + tn * 16 + n;
      if (l >= L_) continue;
      int obase = m0 + wm + tm * 16 + mh;
      #pragma unroll
      for (int r = 0; r < 8; ++r)
        QKV[((size_t)b * OCH + obase + r) * L_ + l] = acc[tm][tn][r];
    }
  }
}

// ---------------- per-channel BN stats over layout [B][CH][per] ----------------
__global__ __launch_bounds__(256) void chan_stats(const float* __restrict__ data,
    const float* __restrict__ g, const float* __restrict__ bb,
    float* __restrict__ sc, float* __restrict__ bi, int CH, int per, int B) {
  const int c = blockIdx.x;
  const int tid = threadIdx.x;
  float s = 0.f, s2 = 0.f;
  const long total = (long)B * per;
  for (long i = tid; i < total; i += 256) {
    long bidx = i / per;
    int  r    = (int)(i - bidx * per);
    float v = data[(bidx * CH + c) * per + r];
    s += v; s2 += v * v;
  }
  __shared__ float rs[256], rs2[256];
  rs[tid] = s; rs2[tid] = s2;
  __syncthreads();
  for (int st = 128; st > 0; st >>= 1) {
    if (tid < st) { rs[tid] += rs[tid + st]; rs2[tid] += rs2[tid + st]; }
    __syncthreads();
  }
  if (tid == 0) {
    float mean = rs[0] / (float)total;
    float var  = rs2[0] / (float)total - mean * mean;
    float scv  = g[c] * rsqrtf(var + EPS_);
    sc[c] = scv;
    bi[c] = bb[c] - mean * scv;
  }
}

// ---------------- similarity logits + partial BN stats ----------------
__global__ __launch_bounds__(256) void sim_kernel(
    const float* __restrict__ qkv, const float* __restrict__ sc, const float* __restrict__ bi,
    const float* __restrict__ q_rpe, const float* __restrict__ k_rpe,
    float* __restrict__ sim, float* __restrict__ sstat) {
  const int h = blockIdx.x;
  const int b = blockIdx.y;
  const int tid = threadIdx.x;
  __shared__ float qn[DK_][L_ + 1];
  __shared__ float kn[DK_][L_ + 1];
  for (int i = tid; i < DK_ * L_; i += 256) {
    int d = i / L_, l = i % L_;
    int cq = h * DK_ + d;
    int ck = KD_ + h * DK_ + d;
    qn[d][l] = qkv[((size_t)b * OCH + cq) * L_ + l] * sc[cq] + bi[cq];
    kn[d][l] = qkv[((size_t)b * OCH + ck) * L_ + l] * sc[ck] + bi[ck];
  }
  __syncthreads();
  float s[3] = {0.f,0.f,0.f}, s2[3] = {0.f,0.f,0.f};
  const size_t chstride = (size_t)H_ * L_ * L_;
  for (int i = tid; i < L_ * L_; i += 256) {
    int l = i / L_, m = i % L_;
    const float* qr = q_rpe + (size_t)(m - l + SPAN - 1) * DK_;
    const float* kr = k_rpe + (size_t)(m - l + SPAN - 1) * DK_;
    float c = 0.f, qs = 0.f, ks = 0.f;
    #pragma unroll 8
    for (int d = 0; d < DK_; ++d) {
      float qv = qn[d][l], kv = kn[d][m];
      c  += qv * kv;
      qs += qv * qr[d];
      ks += kv * kr[d];
    }
    size_t base = (((size_t)b * SIMC + h) * L_ + l) * L_ + m;
    sim[base]                = c;
    sim[base + chstride]     = qs;
    sim[base + 2 * chstride] = ks;
    s[0]+=c;  s2[0]+=c*c;
    s[1]+=qs; s2[1]+=qs*qs;
    s[2]+=ks; s2[2]+=ks*ks;
  }
  #pragma unroll
  for (int g = 0; g < 3; ++g) {
    float v = s[g], v2 = s2[g];
    for (int off = 16; off > 0; off >>= 1) {
      v  += __shfl_down(v,  off);
      v2 += __shfl_down(v2, off);
    }
    if ((tid & 31) == 0) {
      atomicAdd(&sstat[g * H_ + h], v);
      atomicAdd(&sstat[SIMC + g * H_ + h], v2);
    }
  }
}

__global__ void fin_stats(const float* __restrict__ sstat, const float* __restrict__ g,
                          const float* __restrict__ bb, float* __restrict__ sc,
                          float* __restrict__ bi, int nch, float count) {
  int c = blockIdx.x * blockDim.x + threadIdx.x;
  if (c >= nch) return;
  float mean = sstat[c] / count;
  float var  = sstat[nch + c] / count - mean * mean;
  float s = g[c] * rsqrtf(var + EPS_);
  sc[c] = s;
  bi[c] = bb[c] - mean * s;
}

// ---------------- BN(sim)+group-sum+softmax+value retrieval ----------------
__global__ __launch_bounds__(256) void softval_kernel(
    const float* __restrict__ sim, const float* __restrict__ ssc, const float* __restrict__ sbi,
    const float* __restrict__ qkv, const float* __restrict__ qsc, const float* __restrict__ qbi,
    const float* __restrict__ v_rpe, float* __restrict__ outp) {
  const int h = blockIdx.x, b = blockIdx.y, tid = threadIdx.x;
  __shared__ float wgt[L_][L_ + 1];
  __shared__ float vn[DV_][L_ + 1];
  const float sc0 = ssc[h],       bi0 = sbi[h];
  const float sc1 = ssc[H_ + h],  bi1 = sbi[H_ + h];
  const float sc2 = ssc[2*H_+h],  bi2 = sbi[2*H_ + h];
  const size_t chstride = (size_t)H_ * L_ * L_;
  const size_t sbase = ((size_t)b * SIMC + h) * L_ * L_;
  for (int i = tid; i < L_ * L_; i += 256) {
    int l = i / L_, m = i % L_;
    size_t idx = sbase + (size_t)l * L_ + m;
    wgt[l][m] = sim[idx]*sc0 + bi0 + sim[idx+chstride]*sc1 + bi1 + sim[idx+2*chstride]*sc2 + bi2;
  }
  for (int i = tid; i < DV_ * L_; i += 256) {
    int d = i / L_, m = i % L_;
    int cv = 2*KD_ + h * DV_ + d;
    vn[d][m] = qkv[((size_t)b * OCH + cv) * L_ + m] * qsc[cv] + qbi[cv];
  }
  __syncthreads();
  if (tid < L_) {
    int l = tid;
    float mx = -1e30f;
    for (int m = 0; m < L_; ++m) mx = fmaxf(mx, wgt[l][m]);
    float sum = 0.f;
    for (int m = 0; m < L_; ++m) { float e = __expf(wgt[l][m] - mx); wgt[l][m] = e; sum += e; }
    float inv = 1.0f / sum;
    for (int m = 0; m < L_; ++m) wgt[l][m] *= inv;
  }
  __syncthreads();
  for (int i = tid; i < DV_ * L_; i += 256) {
    int d = i % DV_, l = i / DV_;
    float rc = 0.f, rr = 0.f;
    const float* vr = v_rpe + (size_t)(SPAN - 1 - l) * DV_ + d;
    #pragma unroll 8
    for (int m = 0; m < L_; ++m) {
      float wv = wgt[l][m];
      rc += wv * vn[d][m];
      rr += wv * vr[(size_t)m * DV_];
    }
    outp[((size_t)b * OCH + h * DV_ + d) * L_ + l]       = rc;
    outp[((size_t)b * OCH + VD_ + h * DV_ + d) * L_ + l] = rr;
  }
}

// ---------------- final BN + sum of the two halves ----------------
__global__ void combine_kernel(const float* __restrict__ outp, const float* __restrict__ osc,
                               const float* __restrict__ obi, float* __restrict__ out) {
  int i = blockIdx.x * blockDim.x + threadIdx.x;
  const int total = N_ * VD_ * L_;
  if (i >= total) return;
  int l = i % L_;
  int d = (i / L_) % VD_;
  int b = i / (L_ * VD_);
  float v1 = outp[((size_t)b * OCH + d) * L_ + l]        * osc[d]        + obi[d];
  float v2 = outp[((size_t)b * OCH + VD_ + d) * L_ + l]  * osc[VD_ + d]  + obi[VD_ + d];
  out[i] = v1 + v2;
}

extern "C" void kernel_launch(void* const* d_in, const int* in_sizes, int n_in,
                              void* d_out, int out_size, void* d_ws, size_t ws_size,
                              hipStream_t stream) {
  (void)in_sizes; (void)n_in; (void)out_size; (void)ws_size;
  const float* x     = (const float*)d_in[0];
  const float* w_qkv = (const float*)d_in[1];
  const float* bng   = (const float*)d_in[2];
  const float* bnb   = (const float*)d_in[3];
  const float* q_rpe = (const float*)d_in[4];
  const float* k_rpe = (const float*)d_in[5];
  const float* v_rpe = (const float*)d_in[6];
  const float* simg  = (const float*)d_in[7];
  const float* simb  = (const float*)d_in[8];
  const float* outg  = (const float*)d_in[9];
  const float* outb  = (const float*)d_in[10];
  float* out = (float*)d_out;

  char* ws = (char*)d_ws;
  size_t off = 0;
  auto alloc = [&](size_t bytes) -> void* {
    void* p = ws + off;
    off += (bytes + 255) & ~(size_t)255;
    return p;
  };
  float*  qkv   = (float*) alloc((size_t)N_ * OCH * L_ * 4);
  float*  outp  = (float*) alloc((size_t)N_ * OCH * L_ * 4);
  float*  sim   = (float*) alloc((size_t)N_ * SIMC * L_ * L_ * 4);
  __bf16* xb    = (__bf16*)alloc((size_t)N_ * C_ * L_ * 2);
  __bf16* wb    = (__bf16*)alloc((size_t)OCH * C_ * 2);
  float*  qsc   = (float*) alloc(OCH * 4);
  float*  qbi   = (float*) alloc(OCH * 4);
  float*  sstat = (float*) alloc(2 * SIMC * 4);
  float*  ssc   = (float*) alloc(SIMC * 4);
  float*  sbi   = (float*) alloc(SIMC * 4);
  float*  osc   = (float*) alloc(OCH * 4);
  float*  obi   = (float*) alloc(OCH * 4);

  const int nw = OCH * C_;
  cvt_bf16<<<(nw + 255) / 256, 256, 0, stream>>>(w_qkv, wb, nw);
  const int nx = N_ * C_ * L_;
  cvt_bf16<<<(nx + 255) / 256, 256, 0, stream>>>(x, xb, nx);
  zero_kernel<<<1, 64, 0, stream>>>(sstat, 2 * SIMC);

  gemm_qkv<<<dim3(OCH / 256, N_), 256, 0, stream>>>(wb, xb, qkv);
  chan_stats<<<OCH, 256, 0, stream>>>(qkv, bng, bnb, qsc, qbi, OCH, L_, N_);
  sim_kernel<<<dim3(H_, N_), 256, 0, stream>>>(qkv, qsc, qbi, q_rpe, k_rpe, sim, sstat);
  fin_stats<<<1, 32, 0, stream>>>(sstat, simg, simb, ssc, sbi, SIMC, (float)(N_ * L_ * L_));
  softval_kernel<<<dim3(H_, N_), 256, 0, stream>>>(sim, ssc, sbi, qkv, qsc, qbi, v_rpe, outp);
  chan_stats<<<OCH, 256, 0, stream>>>(outp, outg, outb, osc, obi, OCH, L_, N_);
  const int nt = N_ * VD_ * L_;
  combine_kernel<<<(nt + 255) / 256, 256, 0, stream>>>(outp, osc, obi, out);
}
